// CrossAttention_89824946028705
// MI455X (gfx1250) — matrix operand order repaired and checked by reading, not compiled, not run on hardware
//
#include <hip/hip_runtime.h>
#include <hip/hip_bf16.h>

typedef __bf16 bf16_t;
typedef __attribute__((ext_vector_type(16))) __bf16 v16bf;
typedef __attribute__((ext_vector_type(8)))  __bf16 v8bf;
typedef __attribute__((ext_vector_type(8)))  float  v8f;
typedef __attribute__((ext_vector_type(4)))  float  v4f;

#define NB  8
#define SLA 2048
#define SLB 2048
#define SD  512
#define SH  512
#define SCALE 0.04419417382415922f  /* 1/sqrt(512) */

union ABfrag { v16bf v; v8bf h[2]; bf16_t e[16]; };

// ---------------------------------------------------------------------------
// Kernel 1: W[d][h] -> WT[h][d] as bf16 (both Wa and Wb via blockIdx.z)
// ---------------------------------------------------------------------------
__global__ __launch_bounds__(1024)
void wtrans_kernel(const float* __restrict__ Wa, const float* __restrict__ Wb,
                   bf16_t* __restrict__ WaT, bf16_t* __restrict__ WbT) {
  const float* W = blockIdx.z ? Wb : Wa;
  bf16_t* WT = blockIdx.z ? WbT : WaT;
  __shared__ float t[32][33];
  int d0 = blockIdx.x * 32, h0 = blockIdx.y * 32;
  t[threadIdx.y][threadIdx.x] = W[(size_t)(d0 + threadIdx.y) * SH + h0 + threadIdx.x];
  __syncthreads();
  WT[(size_t)(h0 + threadIdx.y) * SD + d0 + threadIdx.x] = (bf16_t)t[threadIdx.x][threadIdx.y];
}

// ---------------------------------------------------------------------------
// Kernel 2: in[b][l][d] -> inT[b][d][l] as bf16 (a and b via blockIdx.z&1)
// ---------------------------------------------------------------------------
__global__ __launch_bounds__(1024)
void intrans_kernel(const float* __restrict__ inA, const float* __restrict__ inB,
                    bf16_t* __restrict__ Ta, bf16_t* __restrict__ Tb) {
  int z = blockIdx.z;
  int b = z >> 1;
  const float* in = (z & 1) ? inB : inA;
  bf16_t* T = (z & 1) ? Tb : Ta;
  __shared__ float t[32][33];
  int l0 = blockIdx.x * 32, d0 = blockIdx.y * 32;
  t[threadIdx.y][threadIdx.x] = in[((size_t)b * SLA + l0 + threadIdx.y) * SD + d0 + threadIdx.x];
  __syncthreads();
  T[((size_t)b * SD + d0 + threadIdx.y) * SLA + l0 + threadIdx.x] = (bf16_t)t[threadIdx.x][threadIdx.y];
}

// ---------------------------------------------------------------------------
// Kernel 3: projection GEMM  mapped[row][h] = (bf16)(in[row][:] @ W + bias)
// rows = NB*SLA flat; per-wave 64x64 tile, 4 waves/WG (128x128 per WG).
// ---------------------------------------------------------------------------
__global__ __launch_bounds__(128)
void proj_kernel(const float* __restrict__ inA, const bf16_t* __restrict__ WaT,
                 const float* __restrict__ ba,
                 const float* __restrict__ inB, const bf16_t* __restrict__ WbT,
                 const float* __restrict__ bb,
                 bf16_t* __restrict__ mapA, bf16_t* __restrict__ mapB) {
  const float* in; const bf16_t* WT; const float* bias; bf16_t* out;
  if (blockIdx.z == 0) { in = inA; WT = WaT; bias = ba; out = mapA; }
  else                 { in = inB; WT = WbT; bias = bb; out = mapB; }

  int lane = threadIdx.x & 31;
  int wave = threadIdx.x >> 5;
  int m0 = blockIdx.x * 128 + (wave >> 1) * 64;
  int n0 = blockIdx.y * 128 + (wave & 1) * 64;
  int half = lane >> 4, lr = lane & 15;
  int klo = half * 8;    // A-fragment K offset
  int kbl = half * 8;    // B-fragment K offset

  v8f acc[4][4] = {};
  for (int kb = 0; kb < SD; kb += 32) {
    ABfrag a[4], bfr[4];
#pragma unroll
    for (int mi = 0; mi < 4; ++mi) {
      const float* ap = in + (size_t)(m0 + mi * 16 + lr) * SD + kb + klo;
      v4f f0 = *(const v4f*)ap;
      v4f f1 = *(const v4f*)(ap + 4);
      v4f f2 = *(const v4f*)(ap + 16);
      v4f f3 = *(const v4f*)(ap + 20);
#pragma unroll
      for (int e = 0; e < 4; ++e) {
        a[mi].e[e]      = (bf16_t)f0[e];
        a[mi].e[4 + e]  = (bf16_t)f1[e];
        a[mi].e[8 + e]  = (bf16_t)f2[e];
        a[mi].e[12 + e] = (bf16_t)f3[e];
      }
    }
#pragma unroll
    for (int ni = 0; ni < 4; ++ni) {
      const bf16_t* bp = WT + (size_t)(n0 + ni * 16 + lr) * SD + kb + kbl;
      bfr[ni].h[0] = *(const v8bf*)bp;
      bfr[ni].h[1] = *(const v8bf*)(bp + 16);
    }
#pragma unroll
    for (int mi = 0; mi < 4; ++mi)
#pragma unroll
      for (int ni = 0; ni < 4; ++ni)
        acc[mi][ni] = __builtin_amdgcn_wmma_f32_16x16x32_bf16(
            false, a[mi].v, false, bfr[ni].v, (short)0, acc[mi][ni], false, false);
  }
  // epilogue: + bias, convert to bf16, store
#pragma unroll
  for (int ni = 0; ni < 4; ++ni) {
    int col = n0 + ni * 16 + lr;
    float bv = bias[col];
#pragma unroll
    for (int mi = 0; mi < 4; ++mi) {
#pragma unroll
      for (int r = 0; r < 8; ++r) {
        int row = m0 + mi * 16 + half * 8 + r;
        out[(size_t)row * SH + col] = (bf16_t)(acc[mi][ni][r] + bv);
      }
    }
  }
}

// ---------------------------------------------------------------------------
// Kernel 4: scores[b][i][j] = mask ? -1e9 : scale * (mapA[i] . mapB[j])
// ---------------------------------------------------------------------------
__global__ __launch_bounds__(128)
void scores_kernel(const bf16_t* __restrict__ mapA, const bf16_t* __restrict__ mapB,
                   const int* __restrict__ maskA, const int* __restrict__ maskB,
                   float* __restrict__ sc) {
  int b = blockIdx.z;
  const bf16_t* mA = mapA + (size_t)b * SLA * SH;
  const bf16_t* mB = mapB + (size_t)b * SLB * SH;
  const int* ma = maskA + (size_t)b * SLA;
  const int* mb = maskB + (size_t)b * SLB;
  float* scb = sc + (size_t)b * SLA * SLB;

  int lane = threadIdx.x & 31;
  int wave = threadIdx.x >> 5;
  int m0 = blockIdx.x * 128 + (wave >> 1) * 64;  // i
  int n0 = blockIdx.y * 128 + (wave & 1) * 64;   // j
  int half = lane >> 4, lr = lane & 15;
  int klo = half * 8, kbl = half * 8;

  v8f acc[4][4] = {};
  for (int kb = 0; kb < SH; kb += 32) {
    ABfrag a[4], bfr[4];
#pragma unroll
    for (int mi = 0; mi < 4; ++mi) {
      const bf16_t* ap = mA + (size_t)(m0 + mi * 16 + lr) * SH + kb + klo;
      a[mi].h[0] = *(const v8bf*)ap;
      a[mi].h[1] = *(const v8bf*)(ap + 16);
    }
#pragma unroll
    for (int ni = 0; ni < 4; ++ni) {
      const bf16_t* bp = mB + (size_t)(n0 + ni * 16 + lr) * SH + kb + kbl;
      bfr[ni].h[0] = *(const v8bf*)bp;
      bfr[ni].h[1] = *(const v8bf*)(bp + 16);
    }
#pragma unroll
    for (int mi = 0; mi < 4; ++mi)
#pragma unroll
      for (int ni = 0; ni < 4; ++ni)
        acc[mi][ni] = __builtin_amdgcn_wmma_f32_16x16x32_bf16(
            false, a[mi].v, false, bfr[ni].v, (short)0, acc[mi][ni], false, false);
  }
#pragma unroll
  for (int ni = 0; ni < 4; ++ni) {
    int j = n0 + ni * 16 + lr;
    int mbj = mb[j];
#pragma unroll
    for (int mi = 0; mi < 4; ++mi) {
#pragma unroll
      for (int r = 0; r < 8; ++r) {
        int i = m0 + mi * 16 + half * 8 + r;
        float v = acc[mi][ni][r] * SCALE;
        if (ma[i] * mbj == 0) v = -1e9f;
        scb[(size_t)i * SLB + j] = v;
      }
    }
  }
}

// ---------------------------------------------------------------------------
// Kernel 5: row softmax stats (over j): rmax[row], rinv[row]=1/sum
// ---------------------------------------------------------------------------
__global__ __launch_bounds__(256)
void rowstats_kernel(const float* __restrict__ sc, float* __restrict__ rmax,
                     float* __restrict__ rinv) {
  int row = blockIdx.x;  // b*SLA + i
  const float* p = sc + (size_t)row * SLB;
  float m = -__builtin_inff(), s = 0.f;
  for (int c = threadIdx.x; c < SLB; c += 256) {
    float x = p[c];
    float nm = fmaxf(m, x);
    s = s * __expf(m - nm) + __expf(x - nm);
    m = nm;
  }
  __shared__ float sm[256], ss[256];
  sm[threadIdx.x] = m; ss[threadIdx.x] = s;
  __syncthreads();
  for (int off = 128; off > 0; off >>= 1) {
    if ((int)threadIdx.x < off) {
      float m1 = sm[threadIdx.x], s1 = ss[threadIdx.x];
      float m2 = sm[threadIdx.x + off], s2 = ss[threadIdx.x + off];
      float nm = fmaxf(m1, m2);
      sm[threadIdx.x] = nm;
      ss[threadIdx.x] = s1 * __expf(m1 - nm) + s2 * __expf(m2 - nm);
    }
    __syncthreads();
  }
  if (threadIdx.x == 0) { rmax[row] = sm[0]; rinv[row] = 1.f / ss[0]; }
}

// ---------------------------------------------------------------------------
// Kernel 6: column softmax stats (over i): each thread owns one column j
// ---------------------------------------------------------------------------
__global__ __launch_bounds__(256)
void colstats_kernel(const float* __restrict__ sc, float* __restrict__ cmax,
                     float* __restrict__ cinv) {
  int b = blockIdx.y;
  int j = blockIdx.x * 256 + threadIdx.x;
  const float* p = sc + (size_t)b * SLA * SLB + j;
  float m = -__builtin_inff(), s = 0.f;
  for (int i = 0; i < SLA; ++i) {
    float x = p[(size_t)i * SLB];
    float nm = fmaxf(m, x);
    s = s * __expf(m - nm) + __expf(x - nm);
    m = nm;
  }
  cmax[(size_t)b * SLB + j] = m;
  cinv[(size_t)b * SLB + j] = 1.f / s;
}

// ---------------------------------------------------------------------------
// Kernel 7: output_a[b][i][d] = input_a[b][i][d] + sum_j P1[i,j]*input_b[j,d]
//           P1[i,j] = exp(s[i,j]-cmax[j]) * cinv[j]   (softmax over axis=1)
// A = exp-transformed score rows (fp32->bf16); B = input_bT bf16 [d][j]
// ---------------------------------------------------------------------------
__global__ __launch_bounds__(128)
void outa_kernel(const float* __restrict__ sc, const float* __restrict__ cmax,
                 const float* __restrict__ cinv, const bf16_t* __restrict__ inTb,
                 const float* __restrict__ inA, float* __restrict__ outA) {
  int b = blockIdx.z;
  const float* scb = sc + (size_t)b * SLA * SLB;
  const float* cm = cmax + (size_t)b * SLB;
  const float* ci = cinv + (size_t)b * SLB;

  int lane = threadIdx.x & 31;
  int wave = threadIdx.x >> 5;
  int m0 = blockIdx.x * 128 + (wave >> 1) * 64;  // i
  int n0 = blockIdx.y * 128 + (wave & 1) * 64;   // d
  int half = lane >> 4, lr = lane & 15;
  int klo = half * 8, kbl = half * 8;

  v8f acc[4][4] = {};
  for (int kb = 0; kb < SLB; kb += 32) {
    int jb = kb + klo;
    v4f cm0 = *(const v4f*)(cm + jb);
    v4f cm1 = *(const v4f*)(cm + jb + 4);
    v4f cm2 = *(const v4f*)(cm + jb + 16);
    v4f cm3 = *(const v4f*)(cm + jb + 20);
    v4f ci0 = *(const v4f*)(ci + jb);
    v4f ci1 = *(const v4f*)(ci + jb + 4);
    v4f ci2 = *(const v4f*)(ci + jb + 16);
    v4f ci3 = *(const v4f*)(ci + jb + 20);
    ABfrag a[4], bfr[4];
#pragma unroll
    for (int mi = 0; mi < 4; ++mi) {
      const float* sp = scb + (size_t)(m0 + mi * 16 + lr) * SLB + jb;
      v4f s0 = *(const v4f*)sp;
      v4f s1 = *(const v4f*)(sp + 4);
      v4f s2 = *(const v4f*)(sp + 16);
      v4f s3 = *(const v4f*)(sp + 20);
#pragma unroll
      for (int e = 0; e < 4; ++e) {
        a[mi].e[e]      = (bf16_t)(__expf(s0[e] - cm0[e]) * ci0[e]);
        a[mi].e[4 + e]  = (bf16_t)(__expf(s1[e] - cm1[e]) * ci1[e]);
        a[mi].e[8 + e]  = (bf16_t)(__expf(s2[e] - cm2[e]) * ci2[e]);
        a[mi].e[12 + e] = (bf16_t)(__expf(s3[e] - cm3[e]) * ci3[e]);
      }
    }
#pragma unroll
    for (int ni = 0; ni < 4; ++ni) {
      const bf16_t* bp = inTb + ((size_t)b * SD + n0 + ni * 16 + lr) * SLB + kb + kbl;
      bfr[ni].h[0] = *(const v8bf*)bp;
      bfr[ni].h[1] = *(const v8bf*)(bp + 16);
    }
#pragma unroll
    for (int mi = 0; mi < 4; ++mi)
#pragma unroll
      for (int ni = 0; ni < 4; ++ni)
        acc[mi][ni] = __builtin_amdgcn_wmma_f32_16x16x32_bf16(
            false, a[mi].v, false, bfr[ni].v, (short)0, acc[mi][ni], false, false);
  }
#pragma unroll
  for (int ni = 0; ni < 4; ++ni) {
    int d = n0 + ni * 16 + lr;
#pragma unroll
    for (int mi = 0; mi < 4; ++mi) {
#pragma unroll
      for (int r = 0; r < 8; ++r) {
        int i = m0 + mi * 16 + half * 8 + r;
        size_t idx = ((size_t)b * SLA + i) * SD + d;
        outA[idx] = acc[mi][ni][r] + inA[idx];
      }
    }
  }
}

// ---------------------------------------------------------------------------
// Kernel 8: output_b[b][j][d] = input_b[b][j][d] + sum_i P2[i,j]*input_a[i,d]
//           P2[i,j] = exp(s[i,j]-rmax[i]) * rinv[i]   (softmax over axis=-1)
// A = transposed exp-transformed scores (M=j, K=i); B = input_aT bf16 [d][i]
// ---------------------------------------------------------------------------
__global__ __launch_bounds__(128)
void outb_kernel(const float* __restrict__ sc, const float* __restrict__ rmax,
                 const float* __restrict__ rinv, const bf16_t* __restrict__ inTa,
                 const float* __restrict__ inB, float* __restrict__ outB) {
  int b = blockIdx.z;
  const float* scb = sc + (size_t)b * SLA * SLB;
  const float* rm = rmax + (size_t)b * SLA;
  const float* ri = rinv + (size_t)b * SLA;

  int lane = threadIdx.x & 31;
  int wave = threadIdx.x >> 5;
  int m0 = blockIdx.x * 128 + (wave >> 1) * 64;  // j
  int n0 = blockIdx.y * 128 + (wave & 1) * 64;   // d
  int half = lane >> 4, lr = lane & 15;
  int klo = half * 8, kbl = half * 8;

  v8f acc[4][4] = {};
  for (int kb = 0; kb < SLA; kb += 32) {
    int ib = kb + klo;
    v4f rm0 = *(const v4f*)(rm + ib);
    v4f rm1 = *(const v4f*)(rm + ib + 4);
    v4f rm2 = *(const v4f*)(rm + ib + 16);
    v4f rm3 = *(const v4f*)(rm + ib + 20);
    v4f ri0 = *(const v4f*)(ri + ib);
    v4f ri1 = *(const v4f*)(ri + ib + 4);
    v4f ri2 = *(const v4f*)(ri + ib + 16);
    v4f ri3 = *(const v4f*)(ri + ib + 20);
    ABfrag a[4], bfr[4];
#pragma unroll
    for (int mi = 0; mi < 4; ++mi) {
      int j = m0 + mi * 16 + lr;
      const float* sp = scb + (size_t)ib * SLB + j;  // scores[ib + e][j], strided
#pragma unroll
      for (int e = 0; e < 4; ++e) {
        float x0 = sp[(size_t)e * SLB];
        float x1 = sp[(size_t)(e + 4) * SLB];
        float x2 = sp[(size_t)(e + 16) * SLB];
        float x3 = sp[(size_t)(e + 20) * SLB];
        a[mi].e[e]      = (bf16_t)(__expf(x0 - rm0[e]) * ri0[e]);
        a[mi].e[4 + e]  = (bf16_t)(__expf(x1 - rm1[e]) * ri1[e]);
        a[mi].e[8 + e]  = (bf16_t)(__expf(x2 - rm2[e]) * ri2[e]);
        a[mi].e[12 + e] = (bf16_t)(__expf(x3 - rm3[e]) * ri3[e]);
      }
    }
#pragma unroll
    for (int ni = 0; ni < 4; ++ni) {
      const bf16_t* bp = inTa + ((size_t)b * SD + n0 + ni * 16 + lr) * SLA + kb + kbl;
      bfr[ni].h[0] = *(const v8bf*)bp;
      bfr[ni].h[1] = *(const v8bf*)(bp + 16);
    }
#pragma unroll
    for (int mi = 0; mi < 4; ++mi)
#pragma unroll
      for (int ni = 0; ni < 4; ++ni)
        acc[mi][ni] = __builtin_amdgcn_wmma_f32_16x16x32_bf16(
            false, a[mi].v, false, bfr[ni].v, (short)0, acc[mi][ni], false, false);
  }
#pragma unroll
  for (int ni = 0; ni < 4; ++ni) {
    int d = n0 + ni * 16 + lr;
#pragma unroll
    for (int mi = 0; mi < 4; ++mi) {
#pragma unroll
      for (int r = 0; r < 8; ++r) {
        int j = m0 + mi * 16 + half * 8 + r;
        size_t idx = ((size_t)b * SLB + j) * SD + d;
        outB[idx] = acc[mi][ni][r] + inB[idx];
      }
    }
  }
}

// ---------------------------------------------------------------------------
// Launch
// ---------------------------------------------------------------------------
extern "C" void kernel_launch(void* const* d_in, const int* in_sizes, int n_in,
                              void* d_out, int out_size, void* d_ws, size_t ws_size,
                              hipStream_t stream) {
  const float* inA  = (const float*)d_in[0];
  const float* inB  = (const float*)d_in[1];
  const int* maskA  = (const int*)d_in[2];
  const int* maskB  = (const int*)d_in[3];
  const float* Wa   = (const float*)d_in[4];
  const float* ba   = (const float*)d_in[5];
  const float* Wb   = (const float*)d_in[6];
  const float* bb   = (const float*)d_in[7];

  float* outA = (float*)d_out;
  float* outB = outA + (size_t)NB * SLA * SD;

  char* ws = (char*)d_ws;
  // offsets (bytes, all 256-aligned)
  bf16_t* WaT  = (bf16_t*)(ws + 0);          //   512 KB
  bf16_t* WbT  = (bf16_t*)(ws + 524288);     //   512 KB
  bf16_t* mapA = (bf16_t*)(ws + 1048576);    //  16 MB
  bf16_t* mapB = (bf16_t*)(ws + 17825792);   //  16 MB
  bf16_t* inTa = (bf16_t*)(ws + 34603008);   //  16 MB
  bf16_t* inTb = (bf16_t*)(ws + 51380224);   //  16 MB
  float*  sc   = (float*)(ws + 68157440);    // 128 MB
  float*  rmax = (float*)(ws + 202375168);
  float*  rinv = rmax + (size_t)NB * SLA;
  float*  cmax = rinv + (size_t)NB * SLA;
  float*  cinv = cmax + (size_t)NB * SLB;

  wtrans_kernel<<<dim3(SD / 32, SH / 32, 2), dim3(32, 32), 0, stream>>>(Wa, Wb, WaT, WbT);
  intrans_kernel<<<dim3(SLA / 32, SD / 32, NB * 2), dim3(32, 32), 0, stream>>>(inA, inB, inTa, inTb);
  proj_kernel<<<dim3(NB * SLA / 128, SH / 128, 2), dim3(128), 0, stream>>>(
      inA, WaT, ba, inB, WbT, bb, mapA, mapB);
  scores_kernel<<<dim3(SLA / 128, SLB / 128, NB), dim3(128), 0, stream>>>(
      mapA, mapB, maskA, maskB, sc);
  rowstats_kernel<<<dim3(NB * SLA), dim3(256), 0, stream>>>(sc, rmax, rinv);
  colstats_kernel<<<dim3(SLB / 256, NB), dim3(256), 0, stream>>>(sc, cmax, cinv);
  outa_kernel<<<dim3(SLA / 128, SD / 128, NB), dim3(128), 0, stream>>>(
      sc, cmax, cinv, inTb, inA, outA);
  outb_kernel<<<dim3(SLB / 128, SD / 128, NB), dim3(128), 0, stream>>>(
      sc, rmax, rinv, inTa, inB, outB);
}
